// MultiBoxLoss_64055142253037
// MI455X (gfx1250) — compile-verified, matrix operand order checked by reading, not run on hardware
//
#include <hip/hip_runtime.h>
#include <math.h>

// ---------------------------------------------------------------------------
// SSD MultiBoxLoss for MI455X (gfx1250, wave32).
//   K1 k_match : per-batch prior<->truth matching, encode, smooth-L1 (pos),
//                conf_t.  One block per batch, LDS-resident best arrays,
//                ds_max_u64 atomics for per-truth argmax over priors.
//   K2 k_ce    : logsumexp over 21 classes via v_wmma_f32_16x16x32_f16
//                (A = exp(x-max) tile, B = ones => row sums), CE, mining.
//   K3 k_select: per-batch radix-select (4x8-bit, LDS histograms) of the
//                num_neg-th largest mining value, exact top-k SUM
//                (sum of k largest is tie-order independent => no sort).
//                Mining array staged into LDS with async global->LDS copies.
//   K4 k_final : serial deterministic reduction of all partials.
// loss_c = sum_{pos} ce + sum(top-num_neg mining)   (mining==0 at pos)
// ---------------------------------------------------------------------------

#define BB 64
#define PP 8732
#define OO 32
#define CC 21

typedef __attribute__((ext_vector_type(16))) _Float16 v16h;
typedef __attribute__((ext_vector_type(8)))  float    v8f;

static constexpr int    CE_BLKX  = (PP + 127) / 128;            // 69
static constexpr size_t WS_LL    = 0;                           // float[64]
static constexpr size_t WS_TOPK  = 256;                         // float[64]
static constexpr size_t WS_NPOS  = 512;                         // int[64]
static constexpr size_t WS_CEP   = 1024;                        // float[69*64]
static constexpr size_t WS_CONFT = 19456;                       // int[BB*PP]
static constexpr size_t WS_MIN   = WS_CONFT + (size_t)BB * PP * 4;

#ifndef __has_builtin
#define __has_builtin(x) 0
#endif

#if __has_builtin(__builtin_amdgcn_global_load_async_to_lds_b32)
#define HAVE_ASYNC_B32 1
#else
#define HAVE_ASYNC_B32 0
#endif

typedef __attribute__((address_space(1))) int gint_t;   // global (__device__)
typedef __attribute__((address_space(3))) int lint_t;   // LDS (__shared__)

__device__ __forceinline__ void wait_async0() {
#if __has_builtin(__builtin_amdgcn_s_wait_asynccnt)
  __builtin_amdgcn_s_wait_asynccnt(0);
#else
  asm volatile("s_wait_asynccnt 0" ::: "memory");
#endif
}

// ---------------------------------------------------------------------------
// Kernel 1: matching + localization loss.  grid = 64 (batch), block = 256.
// ---------------------------------------------------------------------------
__global__ __launch_bounds__(256) void k_match(
    const float* __restrict__ loc, const float* __restrict__ priors,
    const float* __restrict__ targets,
    float* __restrict__ ll_part, int* __restrict__ num_pos_ws,
    int* __restrict__ conf_t_ws)
{
  __shared__ float bt_ov[PP];                 // best truth overlap per prior
  __shared__ unsigned char bt_id[PP];         // best truth index  per prior
  __shared__ unsigned long long bestP[OO];    // per-truth best prior (packed)
  __shared__ float tx1[OO], ty1[OO], tx2[OO], ty2[OO], tar[OO], tlb[OO];
  __shared__ float redf[256];
  __shared__ int   redi[256];

  const int b = blockIdx.x, tid = threadIdx.x;

  if (tid < OO) {
    const float* t = targets + ((size_t)b * OO + tid) * 5;
    float x1 = t[0], y1 = t[1], x2 = t[2], y2 = t[3];
    tx1[tid] = x1; ty1[tid] = y1; tx2[tid] = x2; ty2[tid] = y2;
    tar[tid] = (x2 - x1) * (y2 - y1);
    tlb[tid] = t[4];
    bestP[tid] = 0ull;
  }
  __syncthreads();

  // Phase 1: per-prior best truth; per-truth best prior via ds_max_u64.
  for (int p = tid; p < PP; p += 256) {
    float4 pr = ((const float4*)priors)[p];
    float px1 = pr.x - 0.5f * pr.z, py1 = pr.y - 0.5f * pr.w;
    float px2 = pr.x + 0.5f * pr.z, py2 = pr.y + 0.5f * pr.w;
    float ab  = pr.z * pr.w;
    float bo = -1.0f; int bj = 0;
#pragma unroll 4
    for (int j = 0; j < OO; ++j) {
      float ix = fminf(tx2[j], px2) - fmaxf(tx1[j], px1);
      float iy = fminf(ty2[j], py2) - fmaxf(ty1[j], py1);
      float inter = fmaxf(ix, 0.f) * fmaxf(iy, 0.f);
      float iou = inter / (tar[j] + ab - inter);
      if (iou > bo) { bo = iou; bj = j; }         // argmax(0): first index
      unsigned long long key =
          ((unsigned long long)__float_as_uint(iou) << 32) |
          (unsigned long long)(0xFFFFFFFFu - (unsigned)p);  // ties -> min p
      atomicMax(&bestP[j], key);
    }
    bt_ov[p] = bo;
    bt_id[p] = (unsigned char)bj;
  }
  __syncthreads();

  // Phase 1c: sequential overwrite (matches JAX scatter last-write-wins).
  if (tid == 0) {
    for (int o = 0; o < OO; ++o) {
      unsigned p = 0xFFFFFFFFu - (unsigned)(bestP[o] & 0xFFFFFFFFull);
      bt_ov[p] = 2.0f;
      bt_id[p] = (unsigned char)o;
    }
  }
  __syncthreads();

  // Phase 2: conf_t, encode, smooth-L1 over positives.
  float ll = 0.f; int np = 0;
  for (int p = tid; p < PP; p += 256) {
    int   j  = bt_id[p];
    float ov = bt_ov[p];
    int cls  = (int)tlb[j];
    int ct   = (ov < 0.5f) ? 0 : cls;
    conf_t_ws[(size_t)b * PP + p] = ct;
    if (ct > 0) {
      ++np;
      float4 pr = ((const float4*)priors)[p];
      float mx1 = tx1[j], my1 = ty1[j], mx2 = tx2[j], my2 = ty2[j];
      float g0 = ((mx1 + mx2) * 0.5f - pr.x) / (0.1f * pr.z);
      float g1 = ((my1 + my2) * 0.5f - pr.y) / (0.1f * pr.w);
      float g2 = logf((mx2 - mx1) / pr.z) * 5.0f;   // /VAR1
      float g3 = logf((my2 - my1) / pr.w) * 5.0f;
      float4 l4 = ((const float4*)loc)[(size_t)b * PP + p];
      float d, a;
      d = l4.x - g0; a = fabsf(d); ll += (a < 1.f) ? 0.5f * d * d : a - 0.5f;
      d = l4.y - g1; a = fabsf(d); ll += (a < 1.f) ? 0.5f * d * d : a - 0.5f;
      d = l4.z - g2; a = fabsf(d); ll += (a < 1.f) ? 0.5f * d * d : a - 0.5f;
      d = l4.w - g3; a = fabsf(d); ll += (a < 1.f) ? 0.5f * d * d : a - 0.5f;
    }
  }
  redf[tid] = ll; redi[tid] = np;
  __syncthreads();
  for (int s = 128; s > 0; s >>= 1) {
    if (tid < s) { redf[tid] += redf[tid + s]; redi[tid] += redi[tid + s]; }
    __syncthreads();
  }
  if (tid == 0) { ll_part[b] = redf[0]; num_pos_ws[b] = redi[0]; }
}

// ---------------------------------------------------------------------------
// Kernel 2: CE via WMMA row-sum of exp.  grid = (69, 64), block = 256.
// Each wave handles 16 priors: A[16x32 f16] = exp(x-max) (cols >=21 are 0),
// B = ones  =>  D row m (f32) = sum_c exp(conf[m][c]-max[m]).
// ---------------------------------------------------------------------------
__global__ __launch_bounds__(256) void k_ce(
    const float* __restrict__ conf, const int* __restrict__ conf_t_ws,
    float* __restrict__ mining_ws, float* __restrict__ ce_part)
{
  const int b    = blockIdx.y;
  const int tid  = threadIdx.x;
  const int lane = tid & 31;
  const int wave = tid >> 5;
  const int m    = lane & 15;          // matrix row owned by this lane
  const int h    = lane >> 4;          // half-wave: selects K groups
  const int p0   = (blockIdx.x * 8 + wave) * 16;
  const int p    = p0 + m;
  const bool pv  = (p < PP);

  const float* row = conf + ((size_t)b * PP + (pv ? p : 0)) * CC;
  __builtin_prefetch(row, 0, 1);       // global_prefetch_b8

  // Lane's 16 K-slots: h==0 -> K {0..7,16..23}, h==1 -> K {8..15,24..31}.
  float v[16];
#pragma unroll
  for (int i = 0; i < 16; ++i) {
    int c = (i < 8) ? (8 * h + i) : (8 + 8 * h + i);   // == 16+8h+(i-8)
    v[i] = (pv && c < CC) ? row[c] : -__builtin_inff();
  }
  float pm = v[0];
#pragma unroll
  for (int i = 1; i < 16; ++i) pm = fmaxf(pm, v[i]);
  float fm = fmaxf(pm, __shfl_xor(pm, 16, 32));        // full row max

  v16h a;
#pragma unroll
  for (int i = 0; i < 16; ++i) {
    int c = (i < 8) ? (8 * h + i) : (8 + 8 * h + i);
    float e = (pv && c < CC) ? __expf(v[i] - fm) : 0.0f;
    a[i] = (_Float16)e;
  }
  v16h ones;
#pragma unroll
  for (int i = 0; i < 16; ++i) ones[i] = (_Float16)1.0f;

  v8f acc = {};
  // D = A x Ones + 0  -> every column of row m holds sum_c exp
  acc = __builtin_amdgcn_wmma_f32_16x16x32_f16(
      false, a, false, ones, (short)0, acc, false, false);

  // C/D layout: VGPR r, lanes 0-15 -> M=r ; lanes 16-31 -> M=8+r.
  int r = lane & 7;
  float s = acc[0];
#pragma unroll
  for (int i = 1; i < 8; ++i) if (r == i) s = acc[i];
  int q = (h << 3) | r;                       // prior (within group) we hold
  float qmax = __shfl(fm, q, 32);             // lane q owns prior q's max

  float ce_loc = 0.f;
  if ((lane & 8) == 0) {                      // one writer per prior
    int pq = p0 + q;
    if (pq < PP) {
      int   kq = conf_t_ws[(size_t)b * PP + pq];
      float g  = conf[((size_t)b * PP + pq) * CC + kq];
      float ce = qmax + __logf(s) - g;
      bool pos = kq > 0;
      mining_ws[(size_t)b * PP + pq] = pos ? 0.f : fmaxf(ce, 0.f);
      if (pos) ce_loc = ce;
    }
  }
  __shared__ float redf[256];
  redf[tid] = ce_loc;
  __syncthreads();
  for (int st = 128; st > 0; st >>= 1) {
    if (tid < st) redf[tid] += redf[tid + st];
    __syncthreads();
  }
  if (tid == 0) ce_part[blockIdx.y * gridDim.x + blockIdx.x] = redf[0];
}

// ---------------------------------------------------------------------------
// Kernel 3: per-batch exact sum of the k largest mining values via
// 4-round 8-bit radix select on float bit patterns (all values >= 0).
// ---------------------------------------------------------------------------
__global__ __launch_bounds__(256) void k_select(
    const float* __restrict__ mining_ws, const int* __restrict__ num_pos_ws,
    float* __restrict__ topk_part)
{
  __shared__ float    sm[PP];
  __shared__ unsigned hist[256];
  __shared__ unsigned s_prefix;
  __shared__ int      s_krem;
  __shared__ float    redf[256];
  __shared__ int      redi[256];

  const int b = blockIdx.x, tid = threadIdx.x;
  const float* src = mining_ws + (size_t)b * PP;

#if HAVE_ASYNC_B32
  for (int i = tid; i < PP; i += 256) {
    __builtin_amdgcn_global_load_async_to_lds_b32(
        (gint_t*)(src + i),     // global source (AS1)
        (lint_t*)(&sm[i]),      // LDS destination (AS3)
        0, 0);
  }
  wait_async0();                // this wave's async copies complete
#else
  for (int i = tid; i < PP; i += 256) sm[i] = src[i];
#endif

  int np = num_pos_ws[b];
  int k = 3 * np; if (k > PP - 1) k = PP - 1;
  if (tid == 0) { s_prefix = 0u; s_krem = k; }
  __syncthreads();              // also publishes sm[] across all waves

  if (k > 0) {
    for (int rnd = 0; rnd < 4; ++rnd) {
      int shift = 24 - 8 * rnd;
      unsigned pref = s_prefix;
      hist[tid] = 0u;
      __syncthreads();
      unsigned mask = (rnd == 0) ? 0u : (0xFFFFFFFFu << (32 - 8 * rnd));
      for (int i = tid; i < PP; i += 256) {
        unsigned key = __float_as_uint(sm[i]);
        if ((key & mask) == pref) atomicAdd(&hist[(key >> shift) & 255u], 1u);
      }
      __syncthreads();
      if (tid == 0) {
        unsigned cum = 0; int krem = s_krem;
        for (int bin = 255; bin >= 0; --bin) {
          cum += hist[bin];
          if ((int)cum >= krem) {
            s_krem   = krem - (int)(cum - hist[bin]);
            s_prefix = pref | ((unsigned)bin << shift);
            break;
          }
        }
      }
      __syncthreads();
    }
    unsigned tu = s_prefix;             // exact k-th largest bit pattern
    float    tf = __uint_as_float(tu);
    float sgt = 0.f; int cgt = 0;
    for (int i = tid; i < PP; i += 256) {
      float x = sm[i];
      if (__float_as_uint(x) > tu) { sgt += x; ++cgt; }
    }
    redf[tid] = sgt; redi[tid] = cgt;
    __syncthreads();
    for (int st = 128; st > 0; st >>= 1) {
      if (tid < st) { redf[tid] += redf[tid + st]; redi[tid] += redi[tid + st]; }
      __syncthreads();
    }
    if (tid == 0) topk_part[b] = redf[0] + (float)(k - redi[0]) * tf;
  } else {
    if (tid == 0) topk_part[b] = 0.f;
  }
}

// ---------------------------------------------------------------------------
// Kernel 4: deterministic serial reduction of all partials.
// ---------------------------------------------------------------------------
__global__ void k_final(const float* __restrict__ ll_part,
                        const float* __restrict__ topk_part,
                        const float* __restrict__ ce_part,
                        const int*   __restrict__ num_pos_ws,
                        float* __restrict__ out)
{
  if (threadIdx.x == 0 && blockIdx.x == 0) {
    float ll = 0.f, tk = 0.f, ce = 0.f; int np = 0;
    for (int i = 0; i < BB; ++i) {
      ll += ll_part[i]; tk += topk_part[i]; np += num_pos_ws[i];
    }
    for (int i = 0; i < BB * CE_BLKX; ++i) ce += ce_part[i];
    float N = (float)np; if (N < 1.f) N = 1.f;
    out[0] = (ll + ce + tk) / N;
  }
}

// ---------------------------------------------------------------------------
extern "C" void kernel_launch(void* const* d_in, const int* in_sizes, int n_in,
                              void* d_out, int out_size, void* d_ws, size_t ws_size,
                              hipStream_t stream)
{
  const float* loc     = (const float*)d_in[0];
  const float* conf    = (const float*)d_in[1];
  const float* priors  = (const float*)d_in[2];
  const float* targets = (const float*)d_in[3];

  unsigned char* ws = (unsigned char*)d_ws;
  float* ll_part   = (float*)(ws + WS_LL);
  float* topk_part = (float*)(ws + WS_TOPK);
  int*   num_pos   = (int*)  (ws + WS_NPOS);
  float* ce_part   = (float*)(ws + WS_CEP);
  int*   conf_t    = (int*)  (ws + WS_CONFT);
  float* mining    = (float*)(ws + WS_MIN);

  k_match<<<BB, 256, 0, stream>>>(loc, priors, targets, ll_part, num_pos, conf_t);
  dim3 gce(CE_BLKX, BB);
  k_ce<<<gce, 256, 0, stream>>>(conf, conf_t, mining, ce_part);
  k_select<<<BB, 256, 0, stream>>>(mining, num_pos, topk_part);
  k_final<<<1, 64, 0, stream>>>(ll_part, topk_part, ce_part, num_pos, (float*)d_out);
}
